// ContextBertNER_31413390803683
// MI455X (gfx1250) — matrix operand order repaired and checked by reading, not compiled, not run on hardware
//
#include <hip/hip_runtime.h>
#include <hip/hip_bf16.h>

// ---------------- problem constants ----------------
#define VOCABN 28996
#define TB 32          // B acts as time
#define SN 512         // S acts as LSTM batch
#define DD 768         // hidden size H
#define INN 1536       // LSTM input size 2*D
#define G4 3072        // 4*H
#define NPOS (TB*SN)   // 16384
#define NCLS 9

typedef unsigned short u16;
typedef __attribute__((ext_vector_type(16))) __bf16 v16bf;
typedef __attribute__((ext_vector_type(8)))  float  v8f;

union Frag { v16bf v; uint4 q[2]; };

__device__ __forceinline__ u16 f2bf(float x) {
  unsigned u = __float_as_uint(x);
  unsigned r = u + 0x7FFFu + ((u >> 16) & 1u);   // round-to-nearest-even
  return (u16)(r >> 16);
}
__device__ __forceinline__ float bf2f(u16 s) {
  return __uint_as_float(((unsigned)s) << 16);
}
__device__ __forceinline__ float sigm(float x) { return 1.f / (1.f + __expf(-x)); }

// ---------------- small utility kernels ----------------
__global__ void zero_f32(float* p, long long n) {
  long long i = (long long)blockIdx.x * blockDim.x + threadIdx.x;
  if (i < n) p[i] = 0.f;
}

__global__ void f32_to_bf16(const float* __restrict__ src, u16* __restrict__ dst, long long n) {
  long long i = (long long)blockIdx.x * blockDim.x + threadIdx.x;
  if (i < n) dst[i] = f2bf(src[i]);
}

// segment-sum of embeddings by token id (the "dict loop")
__global__ void seg_accum(const int* __restrict__ ids, const float* __restrict__ hid,
                          float* __restrict__ sums, float* __restrict__ counts) {
  int pos = blockIdx.x;                       // 0..16383
  int id  = ids[pos];
  const float* src = hid + (long long)pos * DD;
  float* dst = sums + (long long)id * DD;
  for (int d = threadIdx.x; d < DD; d += blockDim.x)
    atomicAdd(&dst[d], src[d]);
  if (threadIdx.x == 0) atomicAdd(&counts[id], 1.f);
}

// x = concat(hidden, mean_ctx) converted to bf16, laid out [T=32][N=512][1536]
__global__ void build_x(const int* __restrict__ ids, const float* __restrict__ hid,
                        const float* __restrict__ sums, const float* __restrict__ counts,
                        u16* __restrict__ xb) {
  long long i = (long long)blockIdx.x * blockDim.x + threadIdx.x;
  if (i >= (long long)NPOS * INN) return;
  int pos = (int)(i / INN);
  int c   = (int)(i % INN);
  float v;
  if (c < DD) {
    v = hid[(long long)pos * DD + c];
  } else {
    int id = ids[pos];
    float cnt = counts[id];
    v = sums[(long long)id * DD + (c - DD)] / fmaxf(cnt, 1.f);
  }
  xb[i] = f2bf(v);
}

// ---------------- fused LSTM time-step kernel ----------------
// grid: (24, 16) -> 24 tiles of 32 hidden units x 16 tiles of 32 rows
// block: 128 threads = 4 waves; wave g computes gate g (i,f,g,o) with a
// 2x2 register block (32 rows x 32 cols) -> 2 loads per WMMA instead of 4.
#define KC 128                 // K chunk staged in LDS
#define LDW (KC + 8)           // padded LDS row stride (u16), multiple of 8

__global__ __launch_bounds__(128)
void lstm_step(const u16* __restrict__ xt,    // [512,1536] bf16 (time slice)
               const u16* __restrict__ hprev, // [512,768] bf16 (null if first)
               const u16* __restrict__ Wih,   // [3072,1536] bf16
               const u16* __restrict__ Whh,   // [3072,768]  bf16
               const float* __restrict__ bih, const float* __restrict__ bhh,
               float* __restrict__ Cst,       // [512,768] fp32 cell state
               u16* __restrict__ hout,        // [512,768] bf16
               int first) {
  __shared__ __align__(16) u16  As[32 * LDW]; // A stage: 32 rows x 128 K bf16
  __shared__ float Gs[4][32][33];             // gate exchange, padded

  const int tid   = threadIdx.x;
  const int wave  = tid >> 5;        // gate index 0..3
  const int lane  = tid & 31;
  const int lrow  = lane & 15;       // A row / B column within a 16-tile
  const int khalf = (lane >> 4) * 8; // A fragment K-offset (interleaved layout)
  const int bko   = (lane >> 4) * 16;// unused; B loads also use khalf

  const int colBase = blockIdx.x * 32;   // hidden-unit tile base (0..767)
  const int rowBase = blockIdx.y * 32;   // row tile base (0..511)
  // gate-matrix rows for the two B column tiles
  const long long grow0 = (long long)(wave * DD + colBase +      lrow);
  const long long grow1 = (long long)(wave * DD + colBase + 16 + lrow);

  v8f acc00 = {0,0,0,0,0,0,0,0};   // [row tile][col tile]
  v8f acc01 = {0,0,0,0,0,0,0,0};
  v8f acc10 = {0,0,0,0,0,0,0,0};
  v8f acc11 = {0,0,0,0,0,0,0,0};

  auto gemm_phase = [&](const u16* __restrict__ A, int aStride, int Ktot,
                        const u16* __restrict__ W, int wStride) {
    for (int ks = 0; ks < Ktot; ks += KC) {
      __syncthreads();
      // cooperative stage of A block: 32 rows x 128 K bf16 -> LDS
      #pragma unroll
      for (int i = 0; i < 4; ++i) {
        int e = (tid + i * 128) * 8;   // bf16 element offset in 32x128 block
        int r = e >> 7;                // /128
        int c = e & (KC - 1);
        *(uint4*)&As[r * LDW + c] =
            *(const uint4*)&A[(long long)(rowBase + r) * aStride + ks + c];
      }
      __syncthreads();
      if (ks + KC < Ktot) {            // warm L2->L0 for next weight chunk
        __builtin_prefetch(&W[grow0 * wStride + ks + KC], 0, 1);
        __builtin_prefetch(&W[grow1 * wStride + ks + KC], 0, 1);
      }
      #pragma unroll
      for (int kk = 0; kk < KC; kk += 32) {
        Frag a0, a1, b0, b1;
        // A 16x32 bf16 fragments for the two row tiles
        // (ISA layout: K 0..7 & 16..23 | 8..15 & 24..31 per half-wave)
        a0.q[0] = *(const uint4*)&As[(lrow)      * LDW + kk + khalf];
        a0.q[1] = *(const uint4*)&As[(lrow)      * LDW + kk + khalf + 16];
        a1.q[0] = *(const uint4*)&As[(lrow + 16) * LDW + kk + khalf];
        a1.q[1] = *(const uint4*)&As[(lrow + 16) * LDW + kk + khalf + 16];
        // B 32x16 bf16 fragments: lane = column, K 0..7 & 16..23 | 8..15 & 24..31 per half-wave
        const u16* wp0 = &W[grow0 * wStride + ks + kk + khalf];
        const u16* wp1 = &W[grow1 * wStride + ks + kk + khalf];
        b0.q[0] = *(const uint4*)wp0;
        b0.q[1] = *(const uint4*)(wp0 + 16);
        b1.q[0] = *(const uint4*)wp1;
        b1.q[1] = *(const uint4*)(wp1 + 16);
        acc00 = __builtin_amdgcn_wmma_f32_16x16x32_bf16(
            false, a0.v, false, b0.v, (short)0, acc00, false, false);
        acc01 = __builtin_amdgcn_wmma_f32_16x16x32_bf16(
            false, a0.v, false, b1.v, (short)0, acc01, false, false);
        acc10 = __builtin_amdgcn_wmma_f32_16x16x32_bf16(
            false, a1.v, false, b0.v, (short)0, acc10, false, false);
        acc11 = __builtin_amdgcn_wmma_f32_16x16x32_bf16(
            false, a1.v, false, b1.v, (short)0, acc11, false, false);
      }
    }
  };

  gemm_phase(xt, INN, INN, Wih, INN);             // input projection
  if (!first) gemm_phase(hprev, DD, DD, Whh, DD); // recurrent projection

  // scatter accumulators to LDS: VGPR j, lane l -> row j+(l>=16?8:0), col l%16
  {
    int rb = (lane >> 4) * 8;
    #pragma unroll
    for (int j = 0; j < 8; ++j) {
      Gs[wave][rb + j     ][lrow     ] = acc00[j];
      Gs[wave][rb + j     ][lrow + 16] = acc01[j];
      Gs[wave][rb + j + 16][lrow     ] = acc10[j];
      Gs[wave][rb + j + 16][lrow + 16] = acc11[j];
    }
  }
  __syncthreads();

  // fused gate nonlinearity + cell/hidden update (32x32 outputs per block)
  for (int p = tid; p < 1024; p += 128) {
    int r = p >> 5, c = p & 31;
    int gcol = colBase + c;
    float iv = Gs[0][r][c] + bih[0 * DD + gcol] + bhh[0 * DD + gcol];
    float fv = Gs[1][r][c] + bih[1 * DD + gcol] + bhh[1 * DD + gcol];
    float gv = Gs[2][r][c] + bih[2 * DD + gcol] + bhh[2 * DD + gcol];
    float ov = Gs[3][r][c] + bih[3 * DD + gcol] + bhh[3 * DD + gcol];
    long long idx = (long long)(rowBase + r) * DD + gcol;
    float cprev = first ? 0.f : Cst[idx];
    float cn = sigm(fv) * cprev + sigm(iv) * tanhf(gv);
    float hn = sigm(ov) * tanhf(cn);
    Cst[idx]  = cn;
    hout[idx] = f2bf(hn);
  }
}

// ---------------- final classifier ----------------
__global__ void final_linear(const u16* __restrict__ Hf, const u16* __restrict__ Hb,
                             const float* __restrict__ lw, const float* __restrict__ lb,
                             float* __restrict__ out) {
  int idx = blockIdx.x * blockDim.x + threadIdx.x;   // 16384*9
  if (idx >= NPOS * NCLS) return;
  int pos = idx / NCLS, c = idx % NCLS;
  const u16* hf = Hf + (long long)pos * DD;
  const u16* hb = Hb + (long long)pos * DD;
  const float* w = lw + (long long)c * (2 * DD);
  float s = lb[c];
  for (int k = 0; k < DD; ++k) s += bf2f(hf[k]) * w[k];
  for (int k = 0; k < DD; ++k) s += bf2f(hb[k]) * w[k + DD];
  out[idx] = s;
}

// ---------------- host orchestration ----------------
extern "C" void kernel_launch(void* const* d_in, const int* in_sizes, int n_in,
                              void* d_out, int out_size, void* d_ws, size_t ws_size,
                              hipStream_t stream) {
  const int*   batch  = (const int*)  d_in[0];
  const float* hidden = (const float*)d_in[1];
  const float* w_ih_f = (const float*)d_in[2];
  const float* w_hh_f = (const float*)d_in[3];
  const float* b_ih_f = (const float*)d_in[4];
  const float* b_hh_f = (const float*)d_in[5];
  const float* w_ih_b = (const float*)d_in[6];
  const float* w_hh_b = (const float*)d_in[7];
  const float* b_ih_b = (const float*)d_in[8];
  const float* b_hh_b = (const float*)d_in[9];
  const float* lin_w  = (const float*)d_in[10];
  const float* lin_b  = (const float*)d_in[11];
  float* out = (float*)d_out;

  // deterministic carve-up of workspace
  char* ws = (char*)d_ws;
  auto carve = [&](size_t bytes) {
    char* p = ws;
    ws += (bytes + 255) & ~(size_t)255;
    return p;
  };
  float* sums   = (float*)carve((size_t)VOCABN * DD * 4);
  float* counts = (float*)carve((size_t)VOCABN * 4);
  u16*   xb     = (u16*)  carve((size_t)NPOS * INN * 2);
  u16*   Wihf   = (u16*)  carve((size_t)G4 * INN * 2);
  u16*   Whhf   = (u16*)  carve((size_t)G4 * DD * 2);
  u16*   Wihb   = (u16*)  carve((size_t)G4 * INN * 2);
  u16*   Whhb   = (u16*)  carve((size_t)G4 * DD * 2);
  u16*   Hf     = (u16*)  carve((size_t)NPOS * DD * 2);
  u16*   Hb     = (u16*)  carve((size_t)NPOS * DD * 2);
  float* Cf     = (float*)carve((size_t)SN * DD * 4);
  float* Cb     = (float*)carve((size_t)SN * DD * 4);

  const int TPB = 256;
  // 1) segment mean scratch: zero + accumulate
  {
    long long n = (long long)VOCABN * DD;
    zero_f32<<<(unsigned)((n + TPB - 1) / TPB), TPB, 0, stream>>>(sums, n);
    zero_f32<<<(VOCABN + TPB - 1) / TPB, TPB, 0, stream>>>(counts, VOCABN);
  }
  seg_accum<<<NPOS, TPB, 0, stream>>>(batch, hidden, sums, counts);

  // 2) assemble bf16 x = [hidden | ctx]
  {
    long long n = (long long)NPOS * INN;
    build_x<<<(unsigned)((n + TPB - 1) / TPB), TPB, 0, stream>>>(batch, hidden, sums, counts, xb);
  }

  // 3) weights fp32 -> bf16 (stay resident in 192 MB L2 across all 64 steps)
  {
    long long n1 = (long long)G4 * INN, n2 = (long long)G4 * DD;
    f32_to_bf16<<<(unsigned)((n1 + TPB - 1) / TPB), TPB, 0, stream>>>(w_ih_f, Wihf, n1);
    f32_to_bf16<<<(unsigned)((n2 + TPB - 1) / TPB), TPB, 0, stream>>>(w_hh_f, Whhf, n2);
    f32_to_bf16<<<(unsigned)((n1 + TPB - 1) / TPB), TPB, 0, stream>>>(w_ih_b, Wihb, n1);
    f32_to_bf16<<<(unsigned)((n2 + TPB - 1) / TPB), TPB, 0, stream>>>(w_hh_b, Whhb, n2);
  }

  // 4) bidirectional LSTM: 32 fused WMMA step-kernels per direction
  dim3 sgrid(DD / 32, SN / 32);   // (24, 16)
  for (int t = 0; t < TB; ++t) {
    lstm_step<<<sgrid, 128, 0, stream>>>(
        xb + (size_t)t * SN * INN,
        t ? Hf + (size_t)(t - 1) * SN * DD : nullptr,
        Wihf, Whhf, b_ih_f, b_hh_f, Cf,
        Hf + (size_t)t * SN * DD, t == 0);
  }
  for (int s = 0; s < TB; ++s) {
    int t = TB - 1 - s;                    // reversed time index
    lstm_step<<<sgrid, 128, 0, stream>>>(
        xb + (size_t)t * SN * INN,
        s ? Hb + (size_t)(t + 1) * SN * DD : nullptr,
        Wihb, Whhb, b_ih_b, b_hh_b, Cb,
        Hb + (size_t)t * SN * DD, s == 0);
  }

  // 5) classifier head
  final_linear<<<(NPOS * NCLS + TPB - 1) / TPB, TPB, 0, stream>>>(Hf, Hb, lin_w, lin_b, out);
}
